// SelfAttention_56057913147413
// MI455X (gfx1250) — compile-verified, matrix operand order checked by reading, not run on hardware
//
#include <hip/hip_runtime.h>
#include <hip/hip_bf16.h>
#include <math.h>

// ---------------------------------------------------------------------------
// Problem constants (from reference): B=4, S=1024, HID=1024, NH=16, HD=64
// ---------------------------------------------------------------------------
#define B_      4
#define S_      1024
#define HID_    1024
#define NH_     16
#define HD_     64
#define NEMB_   2047          // 2*MAX_POS - 1
#define NEMB_PAD_ 2048        // one zero row of padding for T-tile overreach

typedef __attribute__((ext_vector_type(16))) __bf16 bf16x16;
typedef __attribute__((ext_vector_type(8)))  __bf16 bf16x8;
typedef __attribute__((ext_vector_type(8)))  float  f32x8;

static __device__ __forceinline__ f32x8 zero8() {
  f32x8 z = {0.f,0.f,0.f,0.f,0.f,0.f,0.f,0.f};
  return z;
}
static __device__ __forceinline__ bf16x16 cat8(bf16x8 lo, bf16x8 hi) {
  return __builtin_shufflevector(lo, hi, 0,1,2,3,4,5,6,7,8,9,10,11,12,13,14,15);
}
static __device__ __forceinline__ f32x8 wmma_bf16(bf16x16 a, bf16x16 b, f32x8 c) {
  // D = A(16x32 bf16) * B(32x16 bf16) + C(16x16 f32)
  return __builtin_amdgcn_wmma_f32_16x16x32_bf16(false, a, false, b,
                                                 (short)0, c, false, false);
}
// Convert 8 contiguous f32 -> 8 bf16 (two float4 loads).
static __device__ __forceinline__ bf16x8 cvt8(const float* p) {
  float4 f0 = *(const float4*)p;
  float4 f1 = *(const float4*)(p + 4);
  bf16x8 r;
  r[0]=(__bf16)f0.x; r[1]=(__bf16)f0.y; r[2]=(__bf16)f0.z; r[3]=(__bf16)f0.w;
  r[4]=(__bf16)f1.x; r[5]=(__bf16)f1.y; r[6]=(__bf16)f1.z; r[7]=(__bf16)f1.w;
  return r;
}
// Load a 16x32 WMMA operand tile where row r (A) / column n (B) of the tile is
// a K-contiguous run in memory at base + r*ld. Per-lane K mapping for 16-bit
// operands (ISA 7.12.2): lane<16 -> K {0..7,16..23}; lane>=16 -> {8..15,24..31}.
static __device__ __forceinline__ bf16x16 load_tile(const __bf16* base, int ld,
                                                    int kbase, int lane) {
  int r  = lane & 15;
  int h8 = (lane >> 4) << 3;
  const __bf16* p = base + (size_t)r * ld + kbase + h8;
  bf16x8 lo = *(const bf16x8*)(p);
  bf16x8 hi = *(const bf16x8*)(p + 16);
  return cat8(lo, hi);
}
// Same, but source is f32 in memory (converted to bf16 on the fly).
static __device__ __forceinline__ bf16x16 load_tile_f32(const float* base, int ld,
                                                        int kbase, int lane) {
  int r  = lane & 15;
  int h8 = (lane >> 4) << 3;
  const float* p = base + (size_t)r * ld + kbase + h8;
  return cat8(cvt8(p), cvt8(p + 16));
}

// ---------------------------------------------------------------------------
// Kernel 0: convert dist_emb f32 -> bf16, with zero pad row (2047 -> 2048 rows)
// ---------------------------------------------------------------------------
__global__ void cvt_dist_kernel(const float* __restrict__ E, __bf16* __restrict__ Eb,
                                int n, int npad) {
  int i = blockIdx.x * 256 + threadIdx.x;
  if (i < npad) Eb[i] = (i < n) ? (__bf16)E[i] : (__bf16)0.0f;
}

// ---------------------------------------------------------------------------
// Kernel 1: fused QKV+CQ projections.  C[m,n] = sum_k X[m,k] * W[n,k] + bias[n]
// One wave -> one 16x16 output tile; K-loop 1024 in steps of 32 (32 WMMAs).
// Q/CQ/K stored bf16 [B,NH,S,HD]; V stored bf16 transposed [B,NH,HD,S].
// ---------------------------------------------------------------------------
__global__ __launch_bounds__(256, 1)
void proj_kernel(const float* __restrict__ X,
                 const float* __restrict__ Wq,  const float* __restrict__ bq,
                 const float* __restrict__ Wk,  const float* __restrict__ bk,
                 const float* __restrict__ Wv,  const float* __restrict__ bv,
                 const float* __restrict__ Wcq, const float* __restrict__ bcq,
                 __bf16* __restrict__ Qb, __bf16* __restrict__ CQb,
                 __bf16* __restrict__ Kb, __bf16* __restrict__ Vt) {
  int lane = threadIdx.x & 31;
  int wid  = threadIdx.x >> 5;
  int w    = blockIdx.x * 8 + wid;          // 65536 wave-tasks total
  int mat  = w >> 14;                        // 0:Q 1:CQ 2:K 3:V
  int rem  = w & 16383;
  int m0   = (rem >> 6) << 4;                // row tile base   [0,4096)
  int n0   = (rem & 63) << 4;                // col tile base   [0,1024)

  const float* W; const float* bias;
  if      (mat == 0) { W = Wq;  bias = bq;  }
  else if (mat == 1) { W = Wcq; bias = bcq; }
  else if (mat == 2) { W = Wk;  bias = bk;  }
  else               { W = Wv;  bias = bv;  }

  int r  = lane & 15;
  int h8 = (lane >> 4) << 3;
  const float* xrow = X + (size_t)(m0 + r) * HID_;
  const float* wrow = W + (size_t)(n0 + r) * HID_;

  f32x8 c = zero8();
  for (int k0 = 0; k0 < HID_; k0 += 32) {
    bf16x16 a  = cat8(cvt8(xrow + k0 + h8), cvt8(xrow + k0 + 16 + h8));
    bf16x16 bb = cat8(cvt8(wrow + k0 + h8), cvt8(wrow + k0 + 16 + h8));
    c = wmma_bf16(a, bb, c);
  }

  int col   = n0 + r;
  float bia = bias[col];
  int hh = col >> 6, d = col & 63;
  #pragma unroll
  for (int v = 0; v < 8; v++) {
    int rowg = m0 + v + h8;                  // global row = b*S + s
    int bi   = rowg >> 10;
    int s    = rowg & 1023;
    __bf16 o = (__bf16)(c[v] + bia);
    if (mat == 3) {
      Vt[(((size_t)(bi * NH_ + hh)) * HD_ + d) * S_ + s] = o;
    } else {
      __bf16* dst = (mat == 0) ? Qb : (mat == 1) ? CQb : Kb;
      dst[(((size_t)(bi * NH_ + hh)) * S_ + s) * HD_ + d] = o;
    }
  }
}

// ---------------------------------------------------------------------------
// Kernel 2: fused attention (flash-style, two score streams + relative term).
// One wave per (b, h, 16-row query tile). Per 32-wide key chunk:
//   4 WMMA (Q·K^T) + 4 (CQ·K^T) + 6 (rel T-tiles) + 4 (P·V) = 18 WMMAs.
// ---------------------------------------------------------------------------
__global__ __launch_bounds__(256, 1)
void attn_kernel(const __bf16* __restrict__ Qb,  const __bf16* __restrict__ CQb,
                 const __bf16* __restrict__ Kb,  const __bf16* __restrict__ Vt,
                 const __bf16* __restrict__ Eb,  const float*  __restrict__ mask,
                 float* __restrict__ out) {
  __shared__ __align__(16) float  Tbuf[8][16 * 48];   // per-wave rel tiles (3 KB)
  __shared__ __align__(16) __bf16 Pbuf[8][16 * 32];   // per-wave probs     (1 KB)

  int lane = threadIdx.x & 31;
  int wid  = threadIdx.x >> 5;
  int task = blockIdx.x * 8 + wid;        // 4096 wave-tasks
  int l0   = (task & 63) << 4;            // query-row tile base
  int h    = (task >> 6) & 15;
  int b    = task >> 10;

  size_t hbase = (size_t)(b * NH_ + h);
  const __bf16* Qh  = Qb  + hbase * S_ * HD_;
  const __bf16* CQh = CQb + hbase * S_ * HD_;
  const __bf16* Kh  = Kb  + hbase * S_ * HD_;
  const __bf16* Vh  = Vt  + hbase * HD_ * S_;
  const float*  mono  = mask + (size_t)b * S_;
  const float*  cross = mask + (size_t)(B_ + b) * S_;

  int cl = lane & 15;                     // tile column owned by this lane
  int h8 = (lane >> 4) << 3;              // row offset of this half-wave

  // Preload A operands (16x64 bf16 each = two 16x32 k-steps).
  bf16x16 aq0 = load_tile(Qh  + (size_t)l0 * HD_, HD_, 0,  lane);
  bf16x16 aq1 = load_tile(Qh  + (size_t)l0 * HD_, HD_, 32, lane);
  bf16x16 ac0 = load_tile(CQh + (size_t)l0 * HD_, HD_, 0,  lane);
  bf16x16 ac1 = load_tile(CQh + (size_t)l0 * HD_, HD_, 32, lane);

  float m_run[8], l_run[8];
  f32x8 acc[4];
  #pragma unroll
  for (int v = 0; v < 8; v++) { m_run[v] = -1e30f; l_run[v] = 0.f; }
  #pragma unroll
  for (int t = 0; t < 4; t++) acc[t] = zero8();

  const float scale = 0.125f;             // 1/sqrt(64)
  float* tb  = Tbuf[wid];
  __bf16* pb = Pbuf[wid];

  for (int r0 = 0; r0 < S_; r0 += 32) {
    // ---- Q·K^T and CQ·K^T (keys r0..r0+31) -------------------------------
    const __bf16* Krow = Kh + (size_t)r0 * HD_;
    if (r0 + 32 < S_) __builtin_prefetch(Krow + 32 * HD_, 0, 1);
    bf16x16 bk00 = load_tile(Krow,            HD_, 0,  lane);
    bf16x16 bk01 = load_tile(Krow,            HD_, 32, lane);
    bf16x16 bk10 = load_tile(Krow + 16 * HD_, HD_, 0,  lane);
    bf16x16 bk11 = load_tile(Krow + 16 * HD_, HD_, 32, lane);

    f32x8 s0 = zero8(), s1 = zero8(), cs0 = zero8(), cs1 = zero8();
    s0  = wmma_bf16(aq0, bk00, s0);  s0  = wmma_bf16(aq1, bk01, s0);
    s1  = wmma_bf16(aq0, bk10, s1);  s1  = wmma_bf16(aq1, bk11, s1);
    cs0 = wmma_bf16(ac0, bk00, cs0); cs0 = wmma_bf16(ac1, bk01, cs0);
    cs1 = wmma_bf16(ac0, bk10, cs1); cs1 = wmma_bf16(ac1, bk11, cs1);

    // ---- relative term: T[i,n] = q[i]·dist_emb[d0+n], n in [0,48) --------
    // needed indices: l0+i - (r0+jj) + 1023, i in [0,16), jj in [0,32)
    int d0 = l0 - r0 + 992;               // window start (always >= 0)
    #pragma unroll
    for (int t = 0; t < 3; t++) {
      const __bf16* Erow = Eb + (size_t)(d0 + t * 16) * HD_;
      bf16x16 be0 = load_tile(Erow, HD_, 0,  lane);
      bf16x16 be1 = load_tile(Erow, HD_, 32, lane);
      f32x8 tt = zero8();
      tt = wmma_bf16(aq0, be0, tt);
      tt = wmma_bf16(aq1, be1, tt);
      #pragma unroll
      for (int v = 0; v < 8; v++)
        tb[(v + h8) * 48 + t * 16 + cl] = tt[v];
    }

    // ---- gather rel, apply scale + masks ---------------------------------
    float mono0  = mono[r0 + cl],       mono1  = mono[r0 + 16 + cl];
    float cross0 = cross[r0 + cl],      cross1 = cross[r0 + 16 + cl];
    #pragma unroll
    for (int v = 0; v < 8; v++) {
      int i = v + h8;
      float rel0 = tb[i * 48 + (i - cl + 31)];        // jj = cl
      float rel1 = tb[i * 48 + (i - (16 + cl) + 31)]; // jj = 16+cl
      s0[v]  = (s0[v] + rel0) * scale + mono0;
      s1[v]  = (s1[v] + rel1) * scale + mono1;
      cs0[v] = cs0[v] * scale + cross0;
      cs1[v] = cs1[v] * scale + cross1;
    }

    // ---- online two-stream softmax update --------------------------------
    float P0[8], P1[8];
    #pragma unroll
    for (int v = 0; v < 8; v++) {
      float mc = fmaxf(fmaxf(s0[v], s1[v]), fmaxf(cs0[v], cs1[v]));
      #pragma unroll
      for (int d = 1; d <= 8; d <<= 1) mc = fmaxf(mc, __shfl_xor(mc, d));
      float mnew = fmaxf(m_run[v], mc);
      float corr = __expf(m_run[v] - mnew);
      m_run[v] = mnew;
      P0[v] = __expf(s0[v] - mnew) + __expf(cs0[v] - mnew);
      P1[v] = __expf(s1[v] - mnew) + __expf(cs1[v] - mnew);
      float rs = P0[v] + P1[v];
      #pragma unroll
      for (int d = 1; d <= 8; d <<= 1) rs += __shfl_xor(rs, d);
      l_run[v] = l_run[v] * corr + rs;
      #pragma unroll
      for (int t = 0; t < 4; t++) acc[t][v] *= corr;
    }

    // ---- P (C-layout) -> LDS -> A-layout bf16 ----------------------------
    #pragma unroll
    for (int v = 0; v < 8; v++) {
      int row = v + h8;
      pb[row * 32 + cl]      = (__bf16)P0[v];
      pb[row * 32 + 16 + cl] = (__bf16)P1[v];
    }
    const __bf16* pr = pb + cl * 32;
    bf16x8 plo = *(const bf16x8*)(pr + h8);
    bf16x8 phi = *(const bf16x8*)(pr + 16 + h8);
    bf16x16 ap = cat8(plo, phi);

    // ---- ctx += P(16x32) · V(32x64)  (V pre-transposed [HD][S]) ----------
    if (r0 + 32 < S_) __builtin_prefetch(Vh + r0 + 32, 0, 1);
    #pragma unroll
    for (int t = 0; t < 4; t++) {
      bf16x16 bv = load_tile(Vh + (size_t)(t * 16) * S_, S_, r0, lane);
      acc[t] = wmma_bf16(ap, bv, acc[t]);
    }
  }

  // ---- epilogue: normalize, write [B,S,HID] f32 --------------------------
  #pragma unroll
  for (int v = 0; v < 8; v++) {
    int row = v + h8;
    float inv = 1.0f / (l_run[v] + 1e-10f);
    size_t obase = ((size_t)(b * S_ + l0 + row)) * HID_ + (size_t)h * HD_;
    #pragma unroll
    for (int t = 0; t < 4; t++)
      out[obase + t * 16 + cl] = acc[t][v] * inv;
  }
}

// ---------------------------------------------------------------------------
// Host launcher
// ---------------------------------------------------------------------------
extern "C" void kernel_launch(void* const* d_in, const int* in_sizes, int n_in,
                              void* d_out, int out_size, void* d_ws, size_t ws_size,
                              hipStream_t stream) {
  const float* X    = (const float*)d_in[0];
  const float* mask = (const float*)d_in[1];
  const float* Wq   = (const float*)d_in[2];  const float* bq  = (const float*)d_in[3];
  const float* Wk   = (const float*)d_in[4];  const float* bk  = (const float*)d_in[5];
  const float* Wv   = (const float*)d_in[6];  const float* bv  = (const float*)d_in[7];
  const float* Wcq  = (const float*)d_in[8];  const float* bcq = (const float*)d_in[9];
  const float* E    = (const float*)d_in[10];
  float* out = (float*)d_out;

  char* ws = (char*)d_ws;
  size_t per = (size_t)B_ * NH_ * S_ * HD_ * sizeof(__bf16);   // 8 MB each
  __bf16* Qb  = (__bf16*)(ws);
  __bf16* CQb = (__bf16*)(ws + per);
  __bf16* Kb  = (__bf16*)(ws + 2 * per);
  __bf16* Vt  = (__bf16*)(ws + 3 * per);
  __bf16* Eb  = (__bf16*)(ws + 4 * per);

  int npad = NEMB_PAD_ * HD_;                                   // 131072
  hipLaunchKernelGGL(cvt_dist_kernel, dim3((npad + 255) / 256), dim3(256), 0, stream,
                     E, Eb, NEMB_ * HD_, npad);

  // 4 matrices * (4096/16) * (1024/16) = 65536 wave-tasks, 8 waves per block
  hipLaunchKernelGGL(proj_kernel, dim3(65536 / 8), dim3(256), 0, stream,
                     X, Wq, bq, Wk, bk, Wv, bv, Wcq, bcq, Qb, CQb, Kb, Vt);

  // B*NH*(S/16) = 4096 wave-tasks, 8 waves per block
  hipLaunchKernelGGL(attn_kernel, dim3(4096 / 8), dim3(256), 0, stream,
                     Qb, CQb, Kb, Vt, Eb, mask, out);
}
